// BudgetAllocationRGNN_65859028517265
// MI455X (gfx1250) — matrix operand order repaired and checked by reading, not compiled, not run on hardware
//
#include <hip/hip_runtime.h>
#include <hip/hip_bf16.h>

// ---------- types for WMMA ----------
typedef __bf16 bf16x16 __attribute__((ext_vector_type(16)));
typedef float  f32x8   __attribute__((ext_vector_type(8)));

union Frag16 {
    bf16x16 v;
    uint4   u[2];   // 2 x 128-bit = 16 bf16
};

__device__ __forceinline__ unsigned short f32_to_bf16_rne(float f) {
    unsigned int u = __float_as_uint(f);
    unsigned int r = u + 0x7FFFu + ((u >> 16) & 1u);
    return (unsigned short)(r >> 16);
}

// ---------- elementwise helpers ----------
__global__ void rgnn_zero_f32(float* p, long long n) {
    long long i = (long long)blockIdx.x * blockDim.x + threadIdx.x;
    if (i < n) p[i] = 0.0f;
}

__global__ void rgnn_f32_to_bf16(const float* __restrict__ s,
                                 unsigned short* __restrict__ d, long long n) {
    long long i = (long long)blockIdx.x * blockDim.x + threadIdx.x;
    if (i < n) d[i] = f32_to_bf16_rne(s[i]);
}

// W[r][k][n] (row-major over n) -> Wt[r][n][k] bf16 (so B fragment loads are contiguous)
__global__ void rgnn_transpose_w_bf16(const float* __restrict__ W,
                                      unsigned short* __restrict__ Wt,
                                      int Rr, int K, int Nn) {
    long long i = (long long)blockIdx.x * blockDim.x + threadIdx.x;
    long long tot = (long long)Rr * K * Nn;
    if (i >= tot) return;
    int n = (int)(i % Nn);
    long long t = i / Nn;
    int k = (int)(t % K);
    int r = (int)(t / K);
    Wt[((long long)r * Nn + n) * K + k] = f32_to_bf16_rne(W[i]);
}

// ---------- edge-side kernels ----------
__global__ void rgnn_count_edges(const int* __restrict__ dst,
                                 const int* __restrict__ et,
                                 float* __restrict__ cnt, int E, int R) {
    int e = blockIdx.x * blockDim.x + threadIdx.x;
    if (e >= E) return;
    atomicAdd(&cnt[(long long)dst[e] * R + et[e]], 1.0f);
}

__global__ void rgnn_recip_norm(float* __restrict__ c, long long n) {
    long long i = (long long)blockIdx.x * blockDim.x + threadIdx.x;
    if (i < n) c[i] = 1.0f / fmaxf(c[i], 1.0f);
}

// one wave (32 lanes) per edge slot; gather h_rel[src,:]*norm -> atomic add into agg[dst,:]
__global__ void rgnn_edge_agg(const int* __restrict__ src,
                              const int* __restrict__ dst,
                              const int* __restrict__ et,
                              const float* __restrict__ norm,
                              const float* __restrict__ hrel,
                              float* __restrict__ agg,
                              int E, int rel, int R, int H) {
    int e = blockIdx.x * blockDim.y + threadIdx.y;
    if (e >= E) return;
    if (et[e] != rel) return;
    int s = src[e], d = dst[e];
    float sc = norm[(long long)d * R + rel];
    const float* hs = hrel + (long long)s * H;
    float* ad = agg + (long long)d * H;
    for (int c = threadIdx.x; c < H; c += 32)
        atomicAdd(&ad[c], hs[c] * sc);
}

// ---------- WMMA GEMM: D = A(bf16,[M,K]) x Bt(bf16,[Nout,K])^T ----------
// Each wave computes a 16x64 output block (4 N-tiles). All 10 fragment loads
// (2xA + 8xB, distinct registers) are issued before the 4 WMMAs so the
// scheduler can clause the loads and use partial loadcnt waits.
// mode 0: outF = A*B            (h_rel, f32)
// mode 1: outH = bf16(relu(A*B + Cin + bias))   (layer-1 output -> layer-2 input)
// mode 2: outF = relu(A*B + Cin + bias)         (layer-2 output, f32)
__global__ void rgnn_wmma_gemm_bf16(const unsigned short* __restrict__ A,
                                    const unsigned short* __restrict__ Bt,
                                    const float* __restrict__ Cin,
                                    const float* __restrict__ bias,
                                    float* __restrict__ outF,
                                    unsigned short* __restrict__ outH,
                                    int M, int K, int Nout, int mode) {
    const int lane   = threadIdx.x;                 // 0..31 (wave32)
    const int tileM  = blockIdx.x * 16;
    const int tileN0 = (blockIdx.y * blockDim.y + threadIdx.y) * 64;  // 4 N-tiles per wave
    if (tileN0 >= Nout) return;                     // wave-uniform exit: EXEC stays all-1s

    const int hi  = lane >> 4;                      // half-wave select
    const int l15 = lane & 15;

    int rowA = tileM + l15; if (rowA >= M) rowA = M - 1;
    const unsigned short* arow = A  + (long long)rowA * K;
    const unsigned short* brow = Bt + (long long)(tileN0 + l15) * K;
    const unsigned short* bp0 = brow;
    const unsigned short* bp1 = brow + (long long)16 * K;
    const unsigned short* bp2 = brow + (long long)32 * K;
    const unsigned short* bp3 = brow + (long long)48 * K;

    f32x8 acc0 = {}, acc1 = {}, acc2 = {}, acc3 = {};
    for (int k = 0; k < K; k += 32) {
        if (k + 32 < K)                             // global_prefetch_b8 for the next A chunk
            __builtin_prefetch(arow + k + 32, 0, 1);

        // A 16x32 bf16 striping: lanes 0-15 hold K = k..k+15;
        // lanes 16-31 hold K = k+16..k+31.
        const int ka = k + hi * 16;
        // B 32x16 bf16 striping (transposed weights, contiguous per lane):
        // lanes 0-15 hold K = k..k+15 of column n, lanes 16-31 hold K = k+16..k+31.
        const int kb = k + hi * 16;

        Frag16 a, b0, b1, b2, b3;                   // distinct live ranges -> clause-able loads
        a.u[0]  = *(const uint4*)(arow + ka);
        a.u[1]  = *(const uint4*)(arow + ka + 8);
        b0.u[0] = *(const uint4*)(bp0 + kb);
        b0.u[1] = *(const uint4*)(bp0 + kb + 8);
        b1.u[0] = *(const uint4*)(bp1 + kb);
        b1.u[1] = *(const uint4*)(bp1 + kb + 8);
        b2.u[0] = *(const uint4*)(bp2 + kb);
        b2.u[1] = *(const uint4*)(bp2 + kb + 8);
        b3.u[0] = *(const uint4*)(bp3 + kb);
        b3.u[1] = *(const uint4*)(bp3 + kb + 8);

        acc0 = __builtin_amdgcn_wmma_f32_16x16x32_bf16(false, a.v, false, b0.v,
                                                       (short)0, acc0, false, false);
        acc1 = __builtin_amdgcn_wmma_f32_16x16x32_bf16(false, a.v, false, b1.v,
                                                       (short)0, acc1, false, false);
        acc2 = __builtin_amdgcn_wmma_f32_16x16x32_bf16(false, a.v, false, b2.v,
                                                       (short)0, acc2, false, false);
        acc3 = __builtin_amdgcn_wmma_f32_16x16x32_bf16(false, a.v, false, b3.v,
                                                       (short)0, acc3, false, false);
    }

    // C/D layout: VGPR j holds (M = tileM + hi*8 + j, N = tileN + l15)
    f32x8 acc[4] = {acc0, acc1, acc2, acc3};
#pragma unroll
    for (int t = 0; t < 4; ++t) {
        const int col = tileN0 + t * 16 + l15;
        if (col >= Nout) continue;
#pragma unroll
        for (int j = 0; j < 8; ++j) {
            int row = tileM + hi * 8 + j;
            if (row >= M) continue;
            long long idx = (long long)row * Nout + col;
            float v = acc[t][j];
            if (mode == 0) {
                outF[idx] = v;
            } else {
                v += Cin[idx] + bias[col];
                v = fmaxf(v, 0.0f);
                if (mode == 1) outH[idx] = f32_to_bf16_rne(v);
                else           outF[idx] = v;
            }
        }
    }
}

// ---------- final scorer: sigmoid(h2 . Ws + bs), one wave per node ----------
__global__ void rgnn_score(const float* __restrict__ h2,
                           const float* __restrict__ Ws,
                           const float* __restrict__ bs,
                           float* __restrict__ out, int Nn, int H) {
    int node = blockIdx.x * blockDim.y + threadIdx.y;
    if (node >= Nn) return;
    int lane = threadIdx.x;
    const float* hp = h2 + (long long)node * H;
    float sum = 0.0f;
    for (int c = lane; c < H; c += 32) sum += hp[c] * Ws[c];
#pragma unroll
    for (int m = 16; m >= 1; m >>= 1) sum += __shfl_xor(sum, m, 32);
    if (lane == 0) out[node] = 1.0f / (1.0f + __expf(-(sum + bs[0])));
}

// ---------- launch ----------
static inline size_t alignup(size_t x) { return (x + 255) & ~(size_t)255; }

extern "C" void kernel_launch(void* const* d_in, const int* in_sizes, int n_in,
                              void* d_out, int out_size, void* d_ws, size_t ws_size,
                              hipStream_t stream) {
    const float* x     = (const float*)d_in[0];
    const int*   eidx  = (const int*)  d_in[1];
    const int*   etype = (const int*)  d_in[2];
    const float* W1    = (const float*)d_in[3];
    const float* root1 = (const float*)d_in[4];
    const float* b1    = (const float*)d_in[5];
    const float* W2    = (const float*)d_in[6];
    const float* root2 = (const float*)d_in[7];
    const float* b2    = (const float*)d_in[8];
    const float* Wsc   = (const float*)d_in[9];
    const float* bsc   = (const float*)d_in[10];

    const int FIN = 128, HID = 256, H2 = 128, R = 8;
    const int E = in_sizes[2];
    const int N = in_sizes[0] / FIN;
    const int* src = eidx;
    const int* dst = eidx + E;

    // workspace carve
    char* p = (char*)d_ws;
    unsigned short* x_bf   = (unsigned short*)p; p += alignup((size_t)N * FIN * 2);
    unsigned short* h1_bf  = (unsigned short*)p; p += alignup((size_t)N * HID * 2);
    unsigned short* w1t    = (unsigned short*)p; p += alignup((size_t)R * HID * FIN * 2);
    unsigned short* r1t    = (unsigned short*)p; p += alignup((size_t)HID * FIN * 2);
    unsigned short* w2t    = (unsigned short*)p; p += alignup((size_t)R * H2 * HID * 2);
    unsigned short* r2t    = (unsigned short*)p; p += alignup((size_t)H2 * HID * 2);
    float* cnt  = (float*)p; p += alignup((size_t)N * R * 4);
    float* hrel = (float*)p; p += alignup((size_t)N * HID * 4);   // reused for layer 2 ([N,H2])
    float* agg1 = (float*)p; p += alignup((size_t)N * HID * 4);
    float* agg2 = (float*)p; p += alignup((size_t)N * H2 * 4);
    float* h2   = (float*)p; p += alignup((size_t)N * H2 * 4);
    float* out  = (float*)d_out;

    const int TB = 256;
    auto blocks = [](long long n, int tb) { return (unsigned)((n + tb - 1) / tb); };

    // 1) counts & norms (identical for both layers)
    rgnn_zero_f32<<<blocks((long long)N * R, TB), TB, 0, stream>>>(cnt, (long long)N * R);
    rgnn_count_edges<<<blocks(E, TB), TB, 0, stream>>>(dst, etype, cnt, E, R);
    rgnn_recip_norm<<<blocks((long long)N * R, TB), TB, 0, stream>>>(cnt, (long long)N * R);

    // 2) precision conversion (activations + transposed weights -> bf16)
    rgnn_f32_to_bf16<<<blocks((long long)N * FIN, TB), TB, 0, stream>>>(x, x_bf, (long long)N * FIN);
    rgnn_transpose_w_bf16<<<blocks((long long)R * FIN * HID, TB), TB, 0, stream>>>(W1, w1t, R, FIN, HID);
    rgnn_transpose_w_bf16<<<blocks((long long)FIN * HID, TB), TB, 0, stream>>>(root1, r1t, 1, FIN, HID);
    rgnn_transpose_w_bf16<<<blocks((long long)R * HID * H2, TB), TB, 0, stream>>>(W2, w2t, R, HID, H2);
    rgnn_transpose_w_bf16<<<blocks((long long)HID * H2, TB), TB, 0, stream>>>(root2, r2t, 1, HID, H2);

    const unsigned mTiles = (unsigned)((N + 15) / 16);
    dim3 gemmBlk1(32, HID / 64);   // 4 waves: 4 x (16x64) covers all 256 columns
    dim3 gemmBlk2(32, H2 / 64);    // 2 waves cover 128 columns
    dim3 gemmGrid(mTiles, 1);
    dim3 edgeBlk(32, 8);
    unsigned edgeGrid = (unsigned)((E + 7) / 8);

    // 3) layer 1: per-relation WMMA transform + mean aggregation
    rgnn_zero_f32<<<blocks((long long)N * HID, TB), TB, 0, stream>>>(agg1, (long long)N * HID);
    for (int r = 0; r < R; ++r) {
        rgnn_wmma_gemm_bf16<<<gemmGrid, gemmBlk1, 0, stream>>>(
            x_bf, w1t + (size_t)r * HID * FIN, nullptr, nullptr,
            hrel, nullptr, N, FIN, HID, 0);
        rgnn_edge_agg<<<edgeGrid, edgeBlk, 0, stream>>>(
            src, dst, etype, cnt, hrel, agg1, E, r, R, HID);
    }
    {   // root transform fused with +agg +bias, ReLU -> bf16 (layer-2 input)
        rgnn_wmma_gemm_bf16<<<gemmGrid, gemmBlk1, 0, stream>>>(
            x_bf, r1t, agg1, b1, nullptr, h1_bf, N, FIN, HID, 1);
    }

    // 4) layer 2
    rgnn_zero_f32<<<blocks((long long)N * H2, TB), TB, 0, stream>>>(agg2, (long long)N * H2);
    for (int r = 0; r < R; ++r) {
        rgnn_wmma_gemm_bf16<<<gemmGrid, gemmBlk2, 0, stream>>>(
            h1_bf, w2t + (size_t)r * H2 * HID, nullptr, nullptr,
            hrel, nullptr, N, HID, H2, 0);
        rgnn_edge_agg<<<edgeGrid, edgeBlk, 0, stream>>>(
            src, dst, etype, cnt, hrel, agg2, E, r, R, H2);
    }
    {   // root transform fused, f32 output for the scorer
        rgnn_wmma_gemm_bf16<<<gemmGrid, gemmBlk2, 0, stream>>>(
            h1_bf, r2t, agg2, b2, h2, nullptr, N, HID, H2, 2);
    }

    // 5) sigmoid(h2 . Ws + bs)
    dim3 scoreBlk(32, 4);
    rgnn_score<<<(unsigned)((N + 3) / 4), scoreBlk, 0, stream>>>(h2, Wsc, bsc, out, N, H2);
}
